// QuantizedLinear_14061722927332
// MI455X (gfx1250) — matrix operand order repaired and checked by reading, not compiled, not run on hardware
//
#include <hip/hip_runtime.h>

typedef __attribute__((ext_vector_type(16))) _Float16     v16h;
typedef __attribute__((ext_vector_type(8)))  _Float16     v8h;
typedef __attribute__((ext_vector_type(8)))  float        v8f;
typedef __attribute__((ext_vector_type(4)))  float        v4f;
typedef __attribute__((ext_vector_type(4)))  int          v4i;
typedef __attribute__((ext_vector_type(8)))  unsigned int v8u;

namespace {
constexpr int M_DIM = 4 * 256;   // 1024 flattened (batch, seq)
constexpr int K_DIM = 4096;
constexpr int N_DIM = 11008;
// fast kernel tiling
constexpr int WM = 128;          // per-wave M (8 x 16 fragments)
constexpr int WN = 64;           // per-wave N (4 x 16 fragments)
constexpr int BM = 256;          // block: 2 waves in M
constexpr int BN = 256;          // block: 4 waves in N
constexpr float MAGIC = 1152.0f; // f16 bit-trick offset: f16(w+1152) = 0x6400|(w+128)
static_assert(M_DIM % BM == 0 && N_DIM % BN == 0 && K_DIM % 32 == 0, "tiling");
}

// ---- pack two int8-valued int32 into one packed-f16 pair equal to (w + 1152) ----
// bits: (lowbyte ^ 0x80) | 0x6400 == lowbyte ^ 0x6480 (no overlap -> OR == XOR)
__device__ __forceinline__ unsigned int pack2_w(int lo, int hi) {
  // v_perm_b32: gather low bytes of {hi,lo} into halves (0x0c selects 0x00)
  return __builtin_amdgcn_perm((unsigned)hi, (unsigned)lo, 0x0c040c00u) ^ 0x64806480u;
}

// Build B fragment (lane = column, K = kh*16 + 0..15 contiguous) from 16 int32.
__device__ __forceinline__ v16h load_bfrag(const int* __restrict__ p, int kb) {
  v4i q0 = *(const v4i*)(p + kb);
  v4i q1 = *(const v4i*)(p + kb + 4);
  v4i q2 = *(const v4i*)(p + kb + 8);
  v4i q3 = *(const v4i*)(p + kb + 12);
  v8u u;
  u[0] = pack2_w(q0[0], q0[1]);  u[1] = pack2_w(q0[2], q0[3]);
  u[2] = pack2_w(q1[0], q1[1]);  u[3] = pack2_w(q1[2], q1[3]);
  u[4] = pack2_w(q2[0], q2[1]);  u[5] = pack2_w(q2[2], q2[3]);
  u[6] = pack2_w(q3[0], q3[1]);  u[7] = pack2_w(q3[2], q3[3]);
  return __builtin_bit_cast(v16h, u);
}

// A fragment from pre-converted f16 x: lane(kh) holds K = kh*16+{0..15} contiguous
__device__ __forceinline__ v16h load_afrag(const _Float16* __restrict__ xrow, int ka) {
  v8h lo = *(const v8h*)(xrow + ka);
  v8h hi = *(const v8h*)(xrow + ka + 8);
  return __builtin_shufflevector(lo, hi, 0,1,2,3,4,5,6,7,8,9,10,11,12,13,14,15);
}

// ===================== pre-pass: x fp32 -> f16, plus rowsum(f16(x)) =====================
__global__ __launch_bounds__(256) void prepass_kernel(
    const float* __restrict__ x, _Float16* __restrict__ xh, float* __restrict__ rs)
{
  const int lane = threadIdx.x & 31;
  const int wave = threadIdx.x >> 5;
  const int row  = blockIdx.x * 8 + wave;            // 128 blocks x 8 rows
  const float*    xr = x  + (size_t)row * K_DIM;
  _Float16*       hr = xh + (size_t)row * K_DIM;
  float sum = 0.0f;
#pragma unroll 4
  for (int it = 0; it < K_DIM; it += 32 * 8) {
    const int base = it + lane * 8;
    v4f a = *(const v4f*)(xr + base);
    v4f b = *(const v4f*)(xr + base + 4);
    v8h h;
#pragma unroll
    for (int i = 0; i < 4; ++i) { h[i] = (_Float16)a[i]; h[4 + i] = (_Float16)b[i]; }
#pragma unroll
    for (int i = 0; i < 8; ++i) sum += (float)h[i];  // rowsum of the ROUNDED values
    *(v8h*)(hr + base) = h;
  }
#pragma unroll
  for (int off = 16; off > 0; off >>= 1) sum += __shfl_xor(sum, off, 32);
  if (lane == 0) rs[row] = sum;
}

// ===================== fast GEMM: 128x64 per wave, 32 WMMAs per K-step =====================
__global__ __launch_bounds__(256) void qlinear_wmma_big(
    const _Float16* __restrict__ xh,    // [M, K] f16 (pre-pass)
    const int*      __restrict__ wq,    // [N, K] int32 (int8-valued)
    const float*    __restrict__ rs,    // [M] rowsum(f16(x))
    const float*    __restrict__ scale_p,
    const float*    __restrict__ zp_p,
    const float*    __restrict__ bias,  // [N]
    float*          __restrict__ out)   // [M, N]
{
  const int lane = threadIdx.x & 31;
  const int wave = threadIdx.x >> 5;
  const int lr   = lane & 15;
  const int kh   = lane >> 4;

  const int m_wave = blockIdx.x * BM + (wave & 1) * WM;
  const int n_wave = blockIdx.y * BN + (wave >> 1) * WN;

  const _Float16* xrow = xh + (size_t)(m_wave + lr) * K_DIM;   // frag mf adds mf*16*K
  const int* w0 = wq + (size_t)(n_wave +  0 + lr) * K_DIM;
  const int* w1 = wq + (size_t)(n_wave + 16 + lr) * K_DIM;
  const int* w2 = wq + (size_t)(n_wave + 32 + lr) * K_DIM;
  const int* w3 = wq + (size_t)(n_wave + 48 + lr) * K_DIM;

  v8f acc[8][4];
#pragma unroll
  for (int mf = 0; mf < 8; ++mf)
#pragma unroll
    for (int j = 0; j < 4; ++j) acc[mf][j] = (v8f){};

  for (int kk = 0; kk < K_DIM; kk += 32) {
    const int kb = kk + kh * 16;
    __builtin_prefetch(w0 + kb + 512, 0, 1);
    __builtin_prefetch(w1 + kb + 512, 0, 1);
    __builtin_prefetch(w2 + kb + 512, 0, 1);
    __builtin_prefetch(w3 + kb + 512, 0, 1);

    v16h bf0 = load_bfrag(w0, kb);
    v16h bf1 = load_bfrag(w1, kb);
    v16h bf2 = load_bfrag(w2, kb);
    v16h bf3 = load_bfrag(w3, kb);

    const int ka = kk + kh * 16;
#pragma unroll
    for (int mf = 0; mf < 8; ++mf) {
      v16h af = load_afrag(xrow + (size_t)mf * 16 * K_DIM, ka);
      acc[mf][0] = __builtin_amdgcn_wmma_f32_16x16x32_f16(false, af, false, bf0,
                                                          (short)0, acc[mf][0], false, false);
      acc[mf][1] = __builtin_amdgcn_wmma_f32_16x16x32_f16(false, af, false, bf1,
                                                          (short)0, acc[mf][1], false, false);
      acc[mf][2] = __builtin_amdgcn_wmma_f32_16x16x32_f16(false, af, false, bf2,
                                                          (short)0, acc[mf][2], false, false);
      acc[mf][3] = __builtin_amdgcn_wmma_f32_16x16x32_f16(false, af, false, bf3,
                                                          (short)0, acc[mf][3], false, false);
    }
  }

  // epilogue: out = scale*acc - scale*(zp + 1152)*rowsum + bias
  const float scale = *scale_p;
  const float c1    = scale * (*zp_p + MAGIC);

#pragma unroll
  for (int mf = 0; mf < 8; ++mf) {
    const int mtop = m_wave + mf * 16 + kh * 8;       // lanes 16-31 hold M+8
    v4f r0 = *(const v4f*)(rs + mtop);
    v4f r1 = *(const v4f*)(rs + mtop + 4);
#pragma unroll
    for (int j = 0; j < 4; ++j) {
      const int n  = n_wave + j * 16 + lr;
      const float bv = bias[n];
      float* orow = out + (size_t)mtop * N_DIM + n;
#pragma unroll
      for (int i = 0; i < 8; ++i) {
        const float rsv = (i < 4) ? r0[i & 3] : r1[i & 3];
        orow[(size_t)i * N_DIM] = scale * acc[mf][j][i] - c1 * rsv + bv;
      }
    }
  }
}

// ===================== fallback (self-contained, ws-free): round-1 kernel =====================
__device__ __forceinline__ v8f wmma_bstep_fb(const int* __restrict__ p, int kb,
                                             v16h af, v8f acc) {
  v4i q0 = *(const v4i*)(p + kb);
  v4i q1 = *(const v4i*)(p + kb + 4);
  v4i q2 = *(const v4i*)(p + kb + 8);
  v4i q3 = *(const v4i*)(p + kb + 12);
  v8u u;
  u[0] = pack2_w(q0[0], q0[1]);  u[1] = pack2_w(q0[2], q0[3]);
  u[2] = pack2_w(q1[0], q1[1]);  u[3] = pack2_w(q1[2], q1[3]);
  u[4] = pack2_w(q2[0], q2[1]);  u[5] = pack2_w(q2[2], q2[3]);
  u[6] = pack2_w(q3[0], q3[1]);  u[7] = pack2_w(q3[2], q3[3]);
  v16h bf = __builtin_bit_cast(v16h, u);
  return __builtin_amdgcn_wmma_f32_16x16x32_f16(false, af, false, bf,
                                                (short)0, acc, false, false);
}

__global__ __launch_bounds__(256, 2) void qlinear_wmma_fallback(
    const float* __restrict__ x, const int* __restrict__ wq,
    const float* __restrict__ scale_p, const float* __restrict__ zp_p,
    const float* __restrict__ bias, float* __restrict__ out)
{
  const int lane = threadIdx.x & 31;
  const int wave = threadIdx.x >> 5;
  const int lr   = lane & 15;
  const int kh   = lane >> 4;

  const int m0 = blockIdx.x * 128 + wave * 16;
  const int n0 = blockIdx.y * 64;

  const float* xrow = x  + (size_t)(m0 + lr) * K_DIM;
  const int*   w0   = wq + (size_t)(n0 +  0 + lr) * K_DIM;
  const int*   w1   = wq + (size_t)(n0 + 16 + lr) * K_DIM;
  const int*   w2   = wq + (size_t)(n0 + 32 + lr) * K_DIM;
  const int*   w3   = wq + (size_t)(n0 + 48 + lr) * K_DIM;

  v16h ones;
#pragma unroll
  for (int i = 0; i < 16; ++i) ones[i] = (_Float16)1.0f;

  v8f acc0 = {}, acc1 = {}, acc2 = {}, acc3 = {}, accS = {};

  for (int kk = 0; kk < K_DIM; kk += 32) {
    const int ka = kk + kh * 16;
    v4f a0 = *(const v4f*)(xrow + ka);
    v4f a1 = *(const v4f*)(xrow + ka + 4);
    v4f a2 = *(const v4f*)(xrow + ka + 8);
    v4f a3 = *(const v4f*)(xrow + ka + 12);
    v16h af;
#pragma unroll
    for (int i = 0; i < 4; ++i) {
      af[i]      = (_Float16)a0[i];
      af[4 + i]  = (_Float16)a1[i];
      af[8 + i]  = (_Float16)a2[i];
      af[12 + i] = (_Float16)a3[i];
    }
    accS = __builtin_amdgcn_wmma_f32_16x16x32_f16(false, af, false, ones,
                                                  (short)0, accS, false, false);
    const int kb = kk + kh * 16;
    acc0 = wmma_bstep_fb(w0, kb, af, acc0);
    acc1 = wmma_bstep_fb(w1, kb, af, acc1);
    acc2 = wmma_bstep_fb(w2, kb, af, acc2);
    acc3 = wmma_bstep_fb(w3, kb, af, acc3);
  }

  const float scale = *scale_p;
  const float c1    = scale * (*zp_p + MAGIC);
  const int   mtop  = m0 + kh * 8;

#pragma unroll
  for (int j = 0; j < 4; ++j) {
    const v8f a = (j == 0) ? acc0 : (j == 1) ? acc1 : (j == 2) ? acc2 : acc3;
    const int n = n0 + j * 16 + lr;
    const float bv = bias[n];
#pragma unroll
    for (int i = 0; i < 8; ++i)
      out[(size_t)(mtop + i) * N_DIM + n] = scale * a[i] - c1 * accS[i] + bv;
  }
}

// ===================== host launcher =====================
extern "C" void kernel_launch(void* const* d_in, const int* in_sizes, int n_in,
                              void* d_out, int out_size, void* d_ws, size_t ws_size,
                              hipStream_t stream) {
  (void)in_sizes; (void)n_in; (void)out_size;
  const float* x       = (const float*)d_in[0];
  const int*   wq      = (const int*)  d_in[1];
  const float* scale_p = (const float*)d_in[2];
  const float* zp_p    = (const float*)d_in[3];
  const float* bias    = (const float*)d_in[4];
  float*       out     = (float*)d_out;

  const size_t xh_bytes = (size_t)M_DIM * K_DIM * sizeof(_Float16);
  const size_t rs_bytes = (size_t)M_DIM * sizeof(float);

  if (ws_size >= xh_bytes + rs_bytes) {
    _Float16* xh = (_Float16*)d_ws;
    float*    rs = (float*)((char*)d_ws + xh_bytes);
    prepass_kernel<<<dim3(M_DIM / 8), dim3(256), 0, stream>>>(x, xh, rs);
    dim3 grid(M_DIM / BM, N_DIM / BN);   // (4, 43): M fastest -> W reuse hits L2
    qlinear_wmma_big<<<grid, dim3(256), 0, stream>>>(xh, wq, rs, scale_p, zp_p, bias, out);
  } else {
    dim3 grid(M_DIM / 128, N_DIM / 64); // (8, 172)
    qlinear_wmma_fallback<<<grid, dim3(256), 0, stream>>>(x, wq, scale_p, zp_p, bias, out);
  }
}
